// RelationOnlyNet_80255758893704
// MI455X (gfx1250) — matrix operand order repaired and checked by reading, not compiled, not run on hardware
//
#include <hip/hip_runtime.h>

#define NPAPER   100000
#define NAUTHOR  50000
#define NSUBJECT 5000
#define NINP     256
#define NHID     128
#define NOUT     16
#define EPA      600000
#define EAP      600000
#define EPS      200000
#define ESP      200000

typedef __attribute__((ext_vector_type(16))) _Float16 v16h;
typedef __attribute__((ext_vector_type(8)))  _Float16 v8h;
typedef __attribute__((ext_vector_type(4)))  _Float16 v4h;
typedef __attribute__((ext_vector_type(8)))  float    v8f;

// -------- pack W[K,N] (f32, row-major) into WMMA B-fragment layout --------
// layout: [kt = K/32][nt = N/16][lane 0..31][16 halves], 32B per lane per tile.
__global__ void __launch_bounds__(256)
pack_w_kernel(const float* __restrict__ W, _Float16* __restrict__ out, int K, int N)
{
    const int ntiles = N >> 4;
    const int total  = (K >> 5) * ntiles * 32;
    const int tid    = blockIdx.x * blockDim.x + threadIdx.x;
    if (tid >= total) return;
    const int lane = tid & 31;
    const int tile = tid >> 5;
    const int kt   = tile / ntiles;
    const int nt   = tile - kt * ntiles;
    const int l15  = lane & 15;
    const int koff = (lane >= 16) ? 8 : 0;
    _Float16*    dst = out + (size_t)tid * 16;
    const float* src = W + (size_t)(kt * 32 + koff) * N + nt * 16 + l15;
#pragma unroll
    for (int h = 0; h < 16; ++h)
        dst[h] = (_Float16)src[(size_t)(h < 8 ? h : h + 8) * N];
}

// -------- fused GEMM: C[M,N] = act(A[M,K] @ W[K,N] + bias), WMMA f16->f32 --------
__global__ void __launch_bounds__(256)
wmma_gemm_bias_act(const float* __restrict__ A, const _Float16* __restrict__ Wf,
                   const float* __restrict__ bias, float* __restrict__ C,
                   int M, int K, int N, int act)
{
    __shared__ _Float16 ldsA[16 * 256];   // 8KB: one 16-row x K(<=256) f16 tile

    const int lane   = threadIdx.x & 31;
    const int wave   = threadIdx.x >> 5;
    const int ntiles = N >> 4;
    const int mtiles = (M + 15) >> 4;
    const int wid    = blockIdx.x * 8 + wave;
    if (wid >= mtiles * ntiles) return;          // uniform per-wave exit

    const int mt = wid / ntiles;
    const int nt = wid - mt * ntiles;
    const int rowBase = mt << 4;
    const int colBase = nt << 4;

    const int  l15    = lane & 15;
    const bool isHi   = lane >= 16;
    const int  aKbase = isHi ? 8 : 0;
    const int  bCol   = colBase + l15;

    const _Float16* bfrag   = Wf + ((size_t)nt * 32 + lane) * 16;
    const size_t    bstride = (size_t)ntiles * 32 * 16;

    v8f c;
#pragma unroll
    for (int i = 0; i < 8; ++i) c[i] = 0.0f;

    if (ntiles == 8) {
        // cooperative stage of the shared 16-row A tile into LDS (f16)
        const int ksh = __popc(K - 1);           // log2(K)
        const int km1 = K - 1;
        const int quads = (16 * K) >> 2;
        for (int q = threadIdx.x; q < quads; q += 256) {
            const int el = q << 2;
            const int r  = el >> ksh;
            const int cc = el & km1;
            v4h h4;
            if (rowBase + r < M) {
                const float4 f = *reinterpret_cast<const float4*>(
                    A + (size_t)(rowBase + r) * K + cc);
                h4[0] = (_Float16)f.x; h4[1] = (_Float16)f.y;
                h4[2] = (_Float16)f.z; h4[3] = (_Float16)f.w;
            } else {
                h4[0] = h4[1] = h4[2] = h4[3] = (_Float16)0.0f;
            }
            *reinterpret_cast<v4h*>(&ldsA[(size_t)r * K + cc]) = h4;
        }
        __syncthreads();

        const _Float16* arow = &ldsA[(size_t)l15 * K];
        for (int kc = 0; kc < K; kc += 32) {
            const v8h alo = *reinterpret_cast<const v8h*>(arow + kc + aKbase);
            const v8h ahi = *reinterpret_cast<const v8h*>(arow + kc + 16 + aKbase);
            v16h a;
#pragma unroll
            for (int i = 0; i < 8; ++i) { a[i] = alo[i]; a[8 + i] = ahi[i]; }
            const v16h b = *reinterpret_cast<const v16h*>(
                bfrag + (size_t)(kc >> 5) * bstride);
            c = __builtin_amdgcn_wmma_f32_16x16x32_f16(
                    false, a, false, b, (short)0, c, false, false);
        }
    } else {
        // direct path (output head, N=16)
        const int  aRow = rowBase + l15;
        const bool aOk  = aRow < M;
        for (int kc = 0; kc < K; kc += 32) {
            float af[16];
            if (aOk) {
                const float* ap = A + (size_t)aRow * K + kc + aKbase;
                const float4 f0 = *reinterpret_cast<const float4*>(ap);
                const float4 f1 = *reinterpret_cast<const float4*>(ap + 4);
                const float4 f2 = *reinterpret_cast<const float4*>(ap + 16);
                const float4 f3 = *reinterpret_cast<const float4*>(ap + 20);
                af[0]=f0.x;  af[1]=f0.y;  af[2]=f0.z;  af[3]=f0.w;
                af[4]=f1.x;  af[5]=f1.y;  af[6]=f1.z;  af[7]=f1.w;
                af[8]=f2.x;  af[9]=f2.y;  af[10]=f2.z; af[11]=f2.w;
                af[12]=f3.x; af[13]=f3.y; af[14]=f3.z; af[15]=f3.w;
            } else {
#pragma unroll
                for (int i = 0; i < 16; ++i) af[i] = 0.0f;
            }
            v16h a;
#pragma unroll
            for (int i = 0; i < 16; ++i) a[i] = (_Float16)af[i];
            const v16h b = *reinterpret_cast<const v16h*>(
                bfrag + (size_t)(kc >> 5) * bstride);
            c = __builtin_amdgcn_wmma_f32_16x16x32_f16(
                    false, a, false, b, (short)0, c, false, false);
        }
    }

    // epilogue: bias + optional tanh-GELU, masked store
    const int   mBase = rowBase + (isHi ? 8 : 0);
    const float bv    = bias ? bias[bCol] : 0.0f;
#pragma unroll
    for (int v = 0; v < 8; ++v) {
        const int row = mBase + v;
        if (row < M) {
            float x = c[v] + bv;
            if (act) {
                const float t = tanhf(0.7978845608028654f * (x + 0.044715f * x * x * x));
                x = 0.5f * x * (1.0f + t);
            }
            C[(size_t)row * N + bCol] = x;
        }
    }
}

// ==================== CSR build (dst-indexed) ====================
__global__ void __launch_bounds__(256)
count_kernel(const int* __restrict__ dst, int* __restrict__ counts, int E)
{
    const int e = blockIdx.x * blockDim.x + threadIdx.x;
    if (e < E) atomicAdd(&counts[dst[e]], 1);
}

// single-workgroup exclusive scan: offsets[0..n] and cursor copy
__global__ void __launch_bounds__(256)
scan_offsets_kernel(const int* __restrict__ counts, int* __restrict__ offsets,
                    int* __restrict__ cursor, int n)
{
    __shared__ int wsum[8];
    __shared__ int wexcl[8];
    __shared__ int running;
    const int lane = threadIdx.x & 31;
    const int wave = threadIdx.x >> 5;
    if (threadIdx.x == 0) running = 0;
    __syncthreads();

    for (int basei = 0; basei < n; basei += 256) {
        const int i = basei + threadIdx.x;
        const int v = (i < n) ? counts[i] : 0;
        // wave-inclusive scan
        int incl = v;
#pragma unroll
        for (int s = 1; s < 32; s <<= 1) {
            const int t = __shfl_up(incl, s, 32);
            if (lane >= s) incl += t;
        }
        if (lane == 31) wsum[wave] = incl;
        __syncthreads();
        if (threadIdx.x == 0) {
            int a = running;
#pragma unroll
            for (int w = 0; w < 8; ++w) { wexcl[w] = a; a += wsum[w]; }
            running = a;
        }
        __syncthreads();
        const int o = wexcl[wave] + incl - v;   // exclusive global offset
        if (i < n) { offsets[i] = o; cursor[i] = o; }
    }
    if (threadIdx.x == 0) offsets[n] = running;
}

__global__ void __launch_bounds__(256)
fill_csr_kernel(const int* __restrict__ src, const int* __restrict__ dst,
                int* __restrict__ cursor, int* __restrict__ csrSrc, int E)
{
    const int e = blockIdx.x * blockDim.x + threadIdx.x;
    if (e < E) {
        const int pos = atomicAdd(&cursor[dst[e]], 1);
        csrSrc[pos] = src[e];
    }
}

// ==================== gather-based segment mean ====================
// One wave per destination node; lane owns 4 features.
// out[d] = beta*out[d] + (sum over incoming edges of t[src]) / max(len,1)
__global__ void __launch_bounds__(256)
gather_mean_kernel(const float* __restrict__ t, const int* __restrict__ csrSrc,
                   const int* __restrict__ offsets, float* __restrict__ out,
                   int n_dst, int beta)
{
    const int lane = threadIdx.x & 31;
    const int d    = blockIdx.x * 8 + (threadIdx.x >> 5);
    if (d >= n_dst) return;
    const int s0 = offsets[d];
    const int s1 = offsets[d + 1];

    float4 a0 = make_float4(0.f, 0.f, 0.f, 0.f);
    float4 a1 = make_float4(0.f, 0.f, 0.f, 0.f);
    int e = s0;
    for (; e + 1 < s1; e += 2) {
        const int sA = csrSrc[e];
        const int sB = csrSrc[e + 1];
        const float4 vA = *reinterpret_cast<const float4*>(t + (size_t)sA * NHID + lane * 4);
        const float4 vB = *reinterpret_cast<const float4*>(t + (size_t)sB * NHID + lane * 4);
        a0.x += vA.x; a0.y += vA.y; a0.z += vA.z; a0.w += vA.w;
        a1.x += vB.x; a1.y += vB.y; a1.z += vB.z; a1.w += vB.w;
    }
    if (e < s1) {
        const int sA = csrSrc[e];
        const float4 vA = *reinterpret_cast<const float4*>(t + (size_t)sA * NHID + lane * 4);
        a0.x += vA.x; a0.y += vA.y; a0.z += vA.z; a0.w += vA.w;
    }
    const float inv = 1.0f / fmaxf((float)(s1 - s0), 1.0f);
    float4 r;
    r.x = (a0.x + a1.x) * inv;
    r.y = (a0.y + a1.y) * inv;
    r.z = (a0.z + a1.z) * inv;
    r.w = (a0.w + a1.w) * inv;

    float* po = out + (size_t)d * NHID + lane * 4;
    if (beta) {
        const float4 o = *reinterpret_cast<const float4*>(po);
        r.x += o.x; r.y += o.y; r.z += o.z; r.w += o.w;
    }
    *reinterpret_cast<float4*>(po) = r;
}

__global__ void __launch_bounds__(256)
fill_zero_kernel(float4* __restrict__ p, int n4)
{
    const int i = blockIdx.x * blockDim.x + threadIdx.x;
    if (i < n4) p[i] = make_float4(0.0f, 0.0f, 0.0f, 0.0f);
}

// ----------------------------------------------------------------------------------
extern "C" void kernel_launch(void* const* d_in, const int* in_sizes, int n_in,
                              void* d_out, int out_size, void* d_ws, size_t ws_size,
                              hipStream_t stream)
{
    (void)in_sizes; (void)n_in; (void)out_size; (void)ws_size;

    const float* x_paper       = (const float*)d_in[0];
    const float* x_author      = (const float*)d_in[1];
    const float* x_subject     = (const float*)d_in[2];
    const int*   pa_src        = (const int*)d_in[3];
    const int*   pa_dst        = (const int*)d_in[4];
    const int*   ap_src        = (const int*)d_in[5];
    const int*   ap_dst        = (const int*)d_in[6];
    const int*   ps_src        = (const int*)d_in[7];
    const int*   ps_dst        = (const int*)d_in[8];
    const int*   sp_src        = (const int*)d_in[9];
    const int*   sp_dst        = (const int*)d_in[10];
    const float* enc_w_paper   = (const float*)d_in[11];
    const float* enc_b_paper   = (const float*)d_in[12];
    const float* enc_w_author  = (const float*)d_in[13];
    const float* enc_b_author  = (const float*)d_in[14];
    const float* enc_w_subject = (const float*)d_in[15];
    const float* enc_b_subject = (const float*)d_in[16];
    const float* wrel[2][4] = {   // order: pa, ap, ps, sp
        {(const float*)d_in[17], (const float*)d_in[18], (const float*)d_in[19], (const float*)d_in[20]},
        {(const float*)d_in[21], (const float*)d_in[22], (const float*)d_in[23], (const float*)d_in[24]}};
    const float* out_w = (const float*)d_in[25];
    const float* out_b = (const float*)d_in[26];

    // ---- workspace carve (256B aligned) ----
    char*  base = (char*)d_ws;
    size_t off  = 0;
    auto carveB = [&](size_t bytes) -> void* {
        void* p = base + off;
        off += (bytes + 255) & ~(size_t)255;
        return p;
    };
    auto carveF = [&](size_t elems) -> float*    { return (float*)carveB(elems * 4); };
    auto carveI = [&](size_t elems) -> int*      { return (int*)carveB(elems * 4); };
    auto carveH = [&](size_t elems) -> _Float16* { return (_Float16*)carveB(elems * 2); };

    float* hp   = carveF((size_t)NPAPER  * NHID);
    float* np_  = carveF((size_t)NPAPER  * NHID);
    float* tbuf = carveF((size_t)NPAPER  * NHID);
    float* ha   = carveF((size_t)NAUTHOR * NHID);
    float* na_  = carveF((size_t)NAUTHOR * NHID);
    float* hs   = carveF((size_t)NSUBJECT* NHID);
    float* ns_  = carveF((size_t)NSUBJECT* NHID);

    // CSR per relation (dst-indexed)
    int* csrPA = carveI(EPA);  int* cntA  = carveI(NAUTHOR);
    int* offA  = carveI(NAUTHOR + 1);  int* curA = carveI(NAUTHOR);
    int* csrPS = carveI(EPS);  int* cntS  = carveI(NSUBJECT);
    int* offS  = carveI(NSUBJECT + 1); int* curS = carveI(NSUBJECT);
    int* csrAP = carveI(EAP);  int* cntPa = carveI(NPAPER);
    int* offPa = carveI(NPAPER + 1);   int* curPa = carveI(NPAPER);
    int* csrSP = carveI(ESP);  int* cntPs = carveI(NPAPER);
    int* offPs = carveI(NPAPER + 1);   int* curPs = carveI(NPAPER);

    // f16 B-fragment buffers
    _Float16* wfEnc[3];
    for (int i = 0; i < 3; ++i) wfEnc[i] = carveH((size_t)NINP * NHID);
    _Float16* wfRel[2][4];
    for (int l = 0; l < 2; ++l)
        for (int r = 0; r < 4; ++r) wfRel[l][r] = carveH((size_t)NHID * NHID);
    _Float16* wfOut = carveH((size_t)NHID * NOUT);

    auto zeroI = [&](int* p, int n) {   // n multiple of 4
        fill_zero_kernel<<<(n / 4 + 255) / 256, 256, 0, stream>>>((float4*)p, n / 4);
    };
    auto build_csr = [&](const int* src, const int* dst, int E, int n_dst,
                         int* counts, int* offsets, int* cursor, int* csrSrc) {
        zeroI(counts, n_dst);
        count_kernel<<<(E + 255) / 256, 256, 0, stream>>>(dst, counts, E);
        scan_offsets_kernel<<<1, 256, 0, stream>>>(counts, offsets, cursor, n_dst);
        fill_csr_kernel<<<(E + 255) / 256, 256, 0, stream>>>(src, dst, cursor, csrSrc, E);
    };
    auto pack = [&](const float* W, _Float16* out, int K, int N) {
        const int total = (K >> 5) * (N >> 4) * 32;
        pack_w_kernel<<<(total + 255) / 256, 256, 0, stream>>>(W, out, K, N);
    };
    auto gemm = [&](const float* A, const _Float16* Wf, const float* b, float* C,
                    int M, int K, int N, int act) {
        const int tiles  = ((M + 15) / 16) * (N / 16);
        const int blocks = (tiles + 7) / 8;
        wmma_gemm_bias_act<<<blocks, 256, 0, stream>>>(A, Wf, b, C, M, K, N, act);
    };
    auto gather = [&](const float* t, const int* csrSrc, const int* offsets,
                      float* out, int n_dst, int beta) {
        gather_mean_kernel<<<(n_dst + 7) / 8, 256, 0, stream>>>(t, csrSrc, offsets, out, n_dst, beta);
    };

    // ---- pack weights into WMMA B-fragment form (f16) ----
    pack(enc_w_paper,   wfEnc[0], NINP, NHID);
    pack(enc_w_author,  wfEnc[1], NINP, NHID);
    pack(enc_w_subject, wfEnc[2], NINP, NHID);
    for (int l = 0; l < 2; ++l)
        for (int r = 0; r < 4; ++r) pack(wrel[l][r], wfRel[l][r], NHID, NHID);
    pack(out_w, wfOut, NHID, NOUT);

    // ---- CSR build (static edge lists; reused by both layers) ----
    build_csr(pa_src, pa_dst, EPA, NAUTHOR,  cntA,  offA,  curA,  csrPA);
    build_csr(ps_src, ps_dst, EPS, NSUBJECT, cntS,  offS,  curS,  csrPS);
    build_csr(ap_src, ap_dst, EAP, NPAPER,   cntPa, offPa, curPa, csrAP);
    build_csr(sp_src, sp_dst, ESP, NPAPER,   cntPs, offPs, curPs, csrSP);

    // ---- encoders: GELU(x @ W + b) ----
    gemm(x_paper,   wfEnc[0], enc_b_paper,   hp, NPAPER,   NINP, NHID, 1);
    gemm(x_author,  wfEnc[1], enc_b_author,  ha, NAUTHOR,  NINP, NHID, 1);
    gemm(x_subject, wfEnc[2], enc_b_subject, hs, NSUBJECT, NINP, NHID, 1);

    // ---- two hetero GraphConv layers: transform -> gather(mean) -> relation sum ----
    float *cp = hp, *ca = ha, *cs = hs;
    float *xp = np_, *xa = na_, *xs = ns_;
    for (int l = 0; l < 2; ++l) {
        gemm(cp, wfRel[l][0], nullptr, tbuf, NPAPER,   NHID, NHID, 0);  // pa
        gather(tbuf, csrPA, offA, xa, NAUTHOR, 0);
        gemm(cp, wfRel[l][2], nullptr, tbuf, NPAPER,   NHID, NHID, 0);  // ps
        gather(tbuf, csrPS, offS, xs, NSUBJECT, 0);
        gemm(ca, wfRel[l][1], nullptr, tbuf, NAUTHOR,  NHID, NHID, 0);  // ap
        gather(tbuf, csrAP, offPa, xp, NPAPER, 0);
        gemm(cs, wfRel[l][3], nullptr, tbuf, NSUBJECT, NHID, NHID, 0);  // sp
        gather(tbuf, csrSP, offPs, xp, NPAPER, 1);                      // += second relation

        float* t;
        t = cp; cp = xp; xp = t;
        t = ca; ca = xa; xa = t;
        t = cs; cs = xs; xs = t;
    }

    // ---- output head ----
    gemm(cp, wfOut, out_b, (float*)d_out, NPAPER, NHID, NOUT, 0);
}
